// GCN_17403207483851
// MI455X (gfx1250) — matrix operand order repaired and checked by reading, not compiled, not run on hardware
//
#include <hip/hip_runtime.h>
#include <hip/hip_bf16.h>
#include <math.h>

#define NN 100000
#define NE 1600000
#define INF 16
#define HID 100
#define OUTF 3
#define KPAD 224      // 2*HID padded to 7*32
#define NPAD 112      // HID padded to 7*16
#define NTILES 7      // column tiles of 16
#define KCHUNKS 7     // K chunks of 32
#define MTILES (NN/16)  // 6250 exactly
#define EGROUPS 25    // float4 groups per edge (25*4 == HID)

typedef __attribute__((ext_vector_type(16))) _Float16 v16h;
typedef __attribute__((ext_vector_type(2)))  _Float16 h2;
typedef __attribute__((ext_vector_type(8)))  float    v8f;
typedef __attribute__((ext_vector_type(4)))  float    f4;
typedef __attribute__((ext_vector_type(2)))  float    f2;

// ---------------- lift: h = tanh(x @ W_lift + b_lift) ----------------
__global__ void lift_kernel(const float* __restrict__ x,
                            const float* __restrict__ Wl,
                            const float* __restrict__ bl,
                            float* __restrict__ h) {
    int t = blockIdx.x * blockDim.x + threadIdx.x;
    if (t >= NN * HID) return;
    int n = t / HID;
    int j = t - n * HID;
    const float* xr = x + (size_t)n * INF;
    float s = bl[j];
#pragma unroll
    for (int k = 0; k < INF; ++k) s += xr[k] * Wl[k * HID + j];
    h[t] = tanhf(s);
}

// ---------------- scatter: reduced[dst] += h[src] * w_e ----------------
// One thread per (edge, float4-group): 25 groups cover HID=100 exactly.
__global__ void scatter_kernel(const float* __restrict__ h,
                               const float* __restrict__ ew,
                               const int* __restrict__ src,
                               const int* __restrict__ dst,
                               float* __restrict__ red) {
    int t = blockIdx.x * blockDim.x + threadIdx.x;
    if (t >= NE * EGROUPS) return;
    int e = t / EGROUPS;
    int g = t - e * EGROUPS;
    int j = g * 4;
    int s = src[e];
    int d = dst[e];
    float w = ew[e];
    f4 v = *(const f4*)(h + (size_t)s * HID + j);
    float* rp = red + (size_t)d * HID + j;
    atomicAdd(rp + 0, v.x * w);
    atomicAdd(rp + 1, v.y * w);
    atomicAdd(rp + 2, v.z * w);
    atomicAdd(rp + 3, v.w * w);
}

// ---------------- prep: W [200x100] f32 -> W16t [112x224] f16 (transposed, padded) ----------------
__global__ void prep_w16(const float* __restrict__ W, _Float16* __restrict__ W16t) {
    int t = blockIdx.x * blockDim.x + threadIdx.x;
    if (t >= NPAD * KPAD) return;
    int n = t / KPAD;
    int k = t - n * KPAD;
    _Float16 v = (_Float16)0.0f;
    if (n < HID && k < 2 * HID) v = (_Float16)W[(size_t)k * HID + n];
    W16t[t] = v;
}

// ---------------- WMMA GEMM: out = relu(concat(h,red) @ W + b) ----------------
// One wave computes a 16-row tile across all 7 column tiles (A reused in regs).
__global__ void mpl_gemm(const float* __restrict__ h,
                         const float* __restrict__ red,
                         const _Float16* __restrict__ W16t,
                         const float* __restrict__ bias,
                         float* __restrict__ out) {
    int wave = threadIdx.x >> 5;
    int lane = threadIdx.x & 31;
    int rowTile = blockIdx.x * (blockDim.x >> 5) + wave;
    if (rowTile >= MTILES) return;   // wave-uniform exit: EXEC stays all-ones for WMMA

    int laneLo = lane & 15;
    int laneHi = lane >> 4;
    int m0 = rowTile * 16;
    const float* hrow = h   + (size_t)(m0 + laneLo) * HID;
    const float* rrow = red + (size_t)(m0 + laneLo) * HID;

    v8f acc[NTILES] = {};

    for (int kc = 0; kc < KCHUNKS; ++kc) {
        // ---- A fragment: 16x32 f16, rows striped per ISA layout ----
        v16h a;
#pragma unroll
        for (int i = 0; i < 8; ++i) {
            int kb = kc * 32 + laneHi * 16 + i * 2;
            f2 v = {};
            if (kb < HID)            v = *(const f2*)(hrow + kb);
            else if (kb < 2 * HID)   v = *(const f2*)(rrow + (kb - HID));
            a[2 * i]     = (_Float16)v.x;
            a[2 * i + 1] = (_Float16)v.y;
        }
        // ---- B fragments for each of the 7 column tiles ----
#pragma unroll
        for (int nt = 0; nt < NTILES; ++nt) {
            int n = nt * 16 + laneLo;
            const _Float16* wrow = W16t + (size_t)n * KPAD + kc * 32 + laneHi * 16;
            v16h b;
#pragma unroll
            for (int i = 0; i < 8; ++i) {
                h2 p = *(const h2*)(wrow + 2 * i);
                b[2 * i]     = p.x;
                b[2 * i + 1] = p.y;
            }
            acc[nt] = __builtin_amdgcn_wmma_f32_16x16x32_f16(
                false, a, false, b, (short)0, acc[nt], false, false);
        }
    }

    // ---- epilogue: bias + ReLU, write only n < 100 ----
#pragma unroll
    for (int nt = 0; nt < NTILES; ++nt) {
        int n = nt * 16 + laneLo;
        if (n < HID) {
            float bv = bias[n];
#pragma unroll
            for (int i = 0; i < 8; ++i) {
                int m = m0 + laneHi * 8 + i;
                float v = acc[nt][i] + bv;
                out[(size_t)m * HID + n] = v > 0.0f ? v : 0.0f;
            }
        }
    }
}

// ---------------- out: sigmoid(h @ W_out + b_out) ----------------
__global__ void out_kernel(const float* __restrict__ h,
                           const float* __restrict__ Wo,
                           const float* __restrict__ bo,
                           float* __restrict__ out) {
    int t = blockIdx.x * blockDim.x + threadIdx.x;
    if (t >= NN * OUTF) return;
    int n = t / OUTF;
    int o = t - n * OUTF;
    const float* hr = h + (size_t)n * HID;
    float s = bo[o];
#pragma unroll 4
    for (int k = 0; k < HID; ++k) s += hr[k] * Wo[k * OUTF + o];
    out[t] = 1.0f / (1.0f + expf(-s));
}

extern "C" void kernel_launch(void* const* d_in, const int* in_sizes, int n_in,
                              void* d_out, int out_size, void* d_ws, size_t ws_size,
                              hipStream_t stream) {
    const float* x   = (const float*)d_in[0];
    const float* ef  = (const float*)d_in[1];
    const int*   src = (const int*)d_in[2];
    const int*   dst = (const int*)d_in[3];
    const float* Wl  = (const float*)d_in[4];
    const float* bl  = (const float*)d_in[5];
    const float* Ws[3] = { (const float*)d_in[6], (const float*)d_in[8], (const float*)d_in[10] };
    const float* bs[3] = { (const float*)d_in[7], (const float*)d_in[9], (const float*)d_in[11] };
    const float* Wo  = (const float*)d_in[12];
    const float* bo  = (const float*)d_in[13];
    float* out = (float*)d_out;

    // workspace layout (~120 MB)
    const size_t nh_bytes = (size_t)NN * HID * sizeof(float);     // 40 MB
    char* ws = (char*)d_ws;
    float* hA  = (float*)ws;                 ws += (nh_bytes + 255) & ~(size_t)255;
    float* hB  = (float*)ws;                 ws += (nh_bytes + 255) & ~(size_t)255;
    float* red = (float*)ws;                 ws += (nh_bytes + 255) & ~(size_t)255;
    _Float16* W16t = (_Float16*)ws;          // 112*224*2 = 50 KB

    // lift
    {
        int total = NN * HID;
        lift_kernel<<<(total + 255) / 256, 256, 0, stream>>>(x, Wl, bl, hA);
    }

    float* hc = hA;
    float* hn = hB;
    for (int l = 0; l < 3; ++l) {
        hipMemsetAsync(red, 0, nh_bytes, stream);
        {
            long long total = (long long)NE * EGROUPS;   // 40M threads
            int grid = (int)((total + 255) / 256);
            scatter_kernel<<<grid, 256, 0, stream>>>(hc, ef, src, dst, red);
        }
        {
            int total = NPAD * KPAD;
            prep_w16<<<(total + 255) / 256, 256, 0, stream>>>(Ws[l], W16t);
        }
        {
            int wavesPerBlock = 256 / 32;
            int grid = (MTILES + wavesPerBlock - 1) / wavesPerBlock;
            mpl_gemm<<<grid, 256, 0, stream>>>(hc, red, W16t, bs[l], hn);
        }
        float* tmp = hc; hc = hn; hn = tmp;
    }

    {
        int total = NN * OUTF;
        out_kernel<<<(total + 255) / 256, 256, 0, stream>>>(hc, Wo, bo, out);
    }
}
